// MultiHeadSelfAttention_17721035063835
// MI455X (gfx1250) — compile-verified, matrix operand order checked by reading, not run on hardware
//
#include <hip/hip_runtime.h>
#include <hip/hip_bf16.h>

typedef __attribute__((ext_vector_type(16))) __bf16 v16bf;
typedef __attribute__((ext_vector_type(8)))  float  v8f;

#define EMBED 1024
#define SEQ   2048
#define BATCH 2
#define HEADS 16
#define HDIM  64
#define ROWS  (BATCH*SEQ)   /* 4096 */

__device__ __forceinline__ v8f wmma_bf16(v16bf a, v16bf b, v8f c) {
  // v_wmma_f32_16x16x32_bf16: D = A(16x32) x B(32x16) + C
  return __builtin_amdgcn_wmma_f32_16x16x32_bf16(false, a, false, b, (short)0, c,
                                                 false, false);
}

// ---- Fragment loaders (ISA 7.12.2 layouts, wave32) --------------------------
// A 16x32 bf16: lane l holds row (l&15); elems 0..7 -> K = kb+e,
// elems 8..15 -> K = 16+kb+e, kb = (l>>4)*8.  Two contiguous 16B runs -> b128.
__device__ __forceinline__ v16bf load_A_bf16(const __bf16* __restrict__ src, int ld,
                                             int row0, int k0, int lane) {
  const __bf16* p = src + (size_t)(row0 + (lane & 15)) * ld + k0 + ((lane >> 4) << 3);
  v16bf a;
#pragma unroll
  for (int e = 0; e < 8; ++e) a[e] = p[e];
#pragma unroll
  for (int e = 0; e < 8; ++e) a[8 + e] = p[16 + e];
  return a;
}

// B 32x16 with B[k][n] = src[(colBase+n)*ld + k0 + k]  (src holds B transposed,
// e.g. Wt[out][in], K^T from K[seq][d], V^T[d][seq]) -> contiguous b128 loads.
__device__ __forceinline__ v16bf load_BT_bf16(const __bf16* __restrict__ src, int ld,
                                              int colBase, int k0, int lane) {
  const __bf16* p = src + (size_t)(colBase + (lane & 15)) * ld + k0 + ((lane >> 4) << 3);
  v16bf b;
#pragma unroll
  for (int e = 0; e < 8; ++e) b[e] = p[e];
#pragma unroll
  for (int e = 0; e < 8; ++e) b[8 + e] = p[16 + e];
  return b;
}

// A fragment out of a row-major 16x32 bf16 LDS tile
__device__ __forceinline__ v16bf load_A_lds(const __bf16* p, int lane) {
  const __bf16* q = p + (lane & 15) * 32 + ((lane >> 4) << 3);
  v16bf a;
#pragma unroll
  for (int e = 0; e < 8; ++e) a[e] = q[e];
#pragma unroll
  for (int e = 0; e < 8; ++e) a[8 + e] = q[16 + e];
  return a;
}

// ---- Pre-pass: x fp32 -> bf16 row-major ------------------------------------
__global__ void __launch_bounds__(256)
cvt_x_kernel(const float* __restrict__ x, __bf16* __restrict__ xb) {
  int i = (blockIdx.x * 256 + threadIdx.x) * 4;
  float4 f = *(const float4*)(x + i);
  xb[i]     = (__bf16)f.x;
  xb[i + 1] = (__bf16)f.y;
  xb[i + 2] = (__bf16)f.z;
  xb[i + 3] = (__bf16)f.w;
}

// ---- Pre-pass: W fp32 [in][out] -> bf16 Wt [out][in] (LDS-tiled transpose) --
__global__ void __launch_bounds__(256)
transpose_w_kernel(const float* __restrict__ W, __bf16* __restrict__ Wt) {
  __shared__ float tile[32][33];
  int tx = threadIdx.x & 31, ty = threadIdx.x >> 5;   // 32 x 8
  int col0 = (blockIdx.x & 31) * 32;
  int row0 = (blockIdx.x >> 5) * 32;
#pragma unroll
  for (int j = 0; j < 4; ++j)
    tile[ty + j * 8][tx] = W[(size_t)(row0 + ty + j * 8) * EMBED + col0 + tx];
  __syncthreads();
#pragma unroll
  for (int j = 0; j < 4; ++j)
    Wt[(size_t)(col0 + ty + j * 8) * EMBED + row0 + tx] = (__bf16)tile[tx][ty + j * 8];
}

// ---- Kernel 1: projection  xb(bf16)[4096x1024] @ Wt^T + b -> bf16 -----------
// Each wave: one 32x64 output block (2 A frags x 4 B tiles -> 8 wmma / k-step;
// B fragment reused across both row tiles, A fragments reused across 4 cols).
// mode 0: dst layout [B][H][N][64] (Q, K);  mode 1: dst layout [B][H][64][N] (V^T)
__global__ void __launch_bounds__(256)
proj_kernel(const __bf16* __restrict__ A, const __bf16* __restrict__ Wt,
            const float* __restrict__ bias, __bf16* __restrict__ dst, int mode) {
  int wave = threadIdx.x >> 5, lane = threadIdx.x & 31;
  int strip = blockIdx.x * 8 + wave;     // 2048 blocks of work: 128 row x 16 col
  int row0 = (strip >> 4) * 32;
  int h = strip & 15;                    // 64-aligned strip == head index
  int col0 = h * 64;

  v8f acc[2][4];
#pragma unroll
  for (int r = 0; r < 2; ++r)
#pragma unroll
    for (int t = 0; t < 4; ++t) acc[r][t] = v8f{};

  for (int k0 = 0; k0 < EMBED; k0 += 32) {
    v16bf a0 = load_A_bf16(A, EMBED, row0,      k0, lane);
    v16bf a1 = load_A_bf16(A, EMBED, row0 + 16, k0, lane);
#pragma unroll
    for (int t = 0; t < 4; ++t) {
      v16bf b = load_BT_bf16(Wt, EMBED, col0 + t * 16, k0, lane);
      acc[0][t] = wmma_bf16(a0, b, acc[0][t]);
      acc[1][t] = wmma_bf16(a1, b, acc[1][t]);
    }
  }
  int n  = lane & 15;
  int mb = (lane >> 4) << 3;
#pragma unroll
  for (int t = 0; t < 4; ++t) {
    int d  = t * 16 + n;
    float bv = bias[col0 + d];
#pragma unroll
    for (int r = 0; r < 2; ++r) {
#pragma unroll
      for (int vv = 0; vv < 8; ++vv) {
        int row = row0 + r * 16 + mb + vv;
        int bi  = row >> 11;               // / SEQ
        int nn  = row & (SEQ - 1);
        __bf16 val = (__bf16)(acc[r][t][vv] + bv);
        size_t base = ((size_t)bi * HEADS + h);
        if (mode == 0)
          dst[(base * SEQ + nn) * HDIM + d] = val;            // [b][h][n][d]
        else
          dst[(base * HDIM + d) * SEQ + nn] = val;            // [b][h][d][n]
      }
    }
  }
}

// ---- Kernel 2: flash attention, one 16-query tile per wave ------------------
__global__ void __launch_bounds__(256)
attention_kernel(const __bf16* __restrict__ q, const __bf16* __restrict__ k,
                 const __bf16* __restrict__ vt, __bf16* __restrict__ attn) {
  __shared__ __bf16 lds_p[8][16 * 32];    // per-wave 16x32 P staging
  int wave = threadIdx.x >> 5, lane = threadIdx.x & 31;
  int tile = blockIdx.x * 8 + wave;       // B*H*(N/16) = 4096 tiles
  int qt = tile & 127;
  int bh = tile >> 7;                     // 0..31
  const __bf16* qb  = q  + (size_t)bh * SEQ * HDIM;
  const __bf16* kb  = k  + (size_t)bh * SEQ * HDIM;
  const __bf16* vtb = vt + (size_t)bh * SEQ * HDIM;   // [64][SEQ]
  int row0 = qt * 16;

  v16bf aq0 = load_A_bf16(qb, HDIM, row0, 0, lane);
  v16bf aq1 = load_A_bf16(qb, HDIM, row0, 32, lane);

  v8f acc[4] = {v8f{}, v8f{}, v8f{}, v8f{}};
  float mrow[8], lrow[8];
#pragma unroll
  for (int vv = 0; vv < 8; ++vv) { mrow[vv] = -3.0e38f; lrow[vv] = 0.0f; }

  const float scale = 0.125f;             // 1/sqrt(64)
  __bf16* pl = &lds_p[wave][0];
  int mb = (lane >> 4) << 3;
  int n  = lane & 15;

  for (int kc = 0; kc < SEQ; kc += 32) {
    if (kc + 32 < SEQ) {                  // prefetch next 32-key K panel (4KB)
      __builtin_prefetch((const char*)(kb + (size_t)(kc + 32) * HDIM) + lane * 128,
                         0, 3);
    }
    v8f s0, s1;
    {
      v16bf b0 = load_BT_bf16(kb, HDIM, kc,      0,  lane);
      v16bf b1 = load_BT_bf16(kb, HDIM, kc,      32, lane);
      v8f c = {};
      c  = wmma_bf16(aq0, b0, c);
      s0 = wmma_bf16(aq1, b1, c);
    }
    {
      v16bf b0 = load_BT_bf16(kb, HDIM, kc + 16, 0,  lane);
      v16bf b1 = load_BT_bf16(kb, HDIM, kc + 16, 32, lane);
      v8f c = {};
      c  = wmma_bf16(aq0, b0, c);
      s1 = wmma_bf16(aq1, b1, c);
    }
    float alpha[8];
#pragma unroll
    for (int vv = 0; vv < 8; ++vv) {
      float sa = s0[vv] * scale, sb = s1[vv] * scale;
      float t = fmaxf(sa, sb);
#pragma unroll
      for (int msk = 1; msk <= 8; msk <<= 1) t = fmaxf(t, __shfl_xor(t, msk, 32));
      float mnew = fmaxf(mrow[vv], t);
      float al = __expf(mrow[vv] - mnew);
      float p0 = __expf(sa - mnew);
      float p1 = __expf(sb - mnew);
      float rs = p0 + p1;
#pragma unroll
      for (int msk = 1; msk <= 8; msk <<= 1) rs += __shfl_xor(rs, msk, 32);
      lrow[vv] = lrow[vv] * al + rs;
      mrow[vv] = mnew;
      alpha[vv] = al;
      int m = mb + vv;
      pl[m * 32 + n]      = (__bf16)p0;
      pl[m * 32 + 16 + n] = (__bf16)p1;
    }
#pragma unroll
    for (int t = 0; t < 4; ++t)
#pragma unroll
      for (int vv = 0; vv < 8; ++vv) acc[t][vv] *= alpha[vv];

    asm volatile("s_wait_dscnt 0" ::: "memory");  // intra-wave LDS RAW fence
    v16bf ap = load_A_lds(pl, lane);
#pragma unroll
    for (int t = 0; t < 4; ++t) {
      // B[k][n=d] = V[key=kc+k][d] = vt[d * SEQ + kc + k] -> contiguous b128
      v16bf bv16 = load_BT_bf16(vtb, SEQ, t * 16, kc, lane);
      acc[t] = wmma_bf16(ap, bv16, acc[t]);
    }
  }

  // normalize and write [B*N][C] bf16 (heads re-interleaved for out proj)
  int bi = bh >> 4, h = bh & 15;
#pragma unroll
  for (int t = 0; t < 4; ++t) {
#pragma unroll
    for (int vv = 0; vv < 8; ++vv) {
      int m = mb + vv;
      float o = acc[t][vv] / lrow[vv];
      attn[(size_t)(bi * SEQ + row0 + m) * EMBED + h * HDIM + t * 16 + n] = (__bf16)o;
    }
  }
}

// ---- Kernel 3: output projection attn(bf16) @ WtO^T + bo -> fp32 ------------
__global__ void __launch_bounds__(256)
out_proj_kernel(const __bf16* __restrict__ A, const __bf16* __restrict__ Wt,
                const float* __restrict__ bias, float* __restrict__ out) {
  int wave = threadIdx.x >> 5, lane = threadIdx.x & 31;
  int strip = blockIdx.x * 8 + wave;     // 128 row x 16 col
  int row0 = (strip >> 4) * 32;
  int col0 = (strip & 15) * 64;

  v8f acc[2][4];
#pragma unroll
  for (int r = 0; r < 2; ++r)
#pragma unroll
    for (int t = 0; t < 4; ++t) acc[r][t] = v8f{};

  for (int k0 = 0; k0 < EMBED; k0 += 32) {
    v16bf a0 = load_A_bf16(A, EMBED, row0,      k0, lane);
    v16bf a1 = load_A_bf16(A, EMBED, row0 + 16, k0, lane);
#pragma unroll
    for (int t = 0; t < 4; ++t) {
      v16bf b = load_BT_bf16(Wt, EMBED, col0 + t * 16, k0, lane);
      acc[0][t] = wmma_bf16(a0, b, acc[0][t]);
      acc[1][t] = wmma_bf16(a1, b, acc[1][t]);
    }
  }
  int n  = lane & 15;
  int mb = (lane >> 4) << 3;
#pragma unroll
  for (int t = 0; t < 4; ++t) {
    int col = col0 + t * 16 + n;
    float bv = bias[col];
#pragma unroll
    for (int r = 0; r < 2; ++r)
#pragma unroll
      for (int vv = 0; vv < 8; ++vv)
        out[(size_t)(row0 + r * 16 + mb + vv) * EMBED + col] = acc[r][t][vv] + bv;
  }
}

extern "C" void kernel_launch(void* const* d_in, const int* in_sizes, int n_in,
                              void* d_out, int out_size, void* d_ws, size_t ws_size,
                              hipStream_t stream) {
  const float* x  = (const float*)d_in[0];
  const float* Wq = (const float*)d_in[1];
  const float* bq = (const float*)d_in[2];
  const float* Wk = (const float*)d_in[3];
  const float* bk = (const float*)d_in[4];
  const float* Wv = (const float*)d_in[5];
  const float* bv = (const float*)d_in[6];
  const float* Wo = (const float*)d_in[7];
  const float* bo = (const float*)d_in[8];
  float* out = (float*)d_out;

  const size_t XN = (size_t)ROWS * EMBED;          // 4,194,304
  const size_t WN = (size_t)EMBED * EMBED;         // 1,048,576
  __bf16* xb   = (__bf16*)d_ws;
  __bf16* wtq  = xb  + XN;
  __bf16* wtk  = wtq + WN;
  __bf16* wtv  = wtk + WN;
  __bf16* wto  = wtv + WN;
  __bf16* qws  = wto + WN;                         // [B][H][N][64]
  __bf16* kws  = qws + XN;                         // [B][H][N][64]
  __bf16* vtws = kws + XN;                         // [B][H][64][N]
  __bf16* aws  = vtws + XN;                        // [4096][1024]
  // total: 5*XN + 4*WN bf16 = 48 MB

  dim3 blk(256);
  cvt_x_kernel      <<<dim3(XN / 1024), blk, 0, stream>>>(x, xb);
  transpose_w_kernel<<<dim3(1024), blk, 0, stream>>>(Wq, wtq);
  transpose_w_kernel<<<dim3(1024), blk, 0, stream>>>(Wk, wtk);
  transpose_w_kernel<<<dim3(1024), blk, 0, stream>>>(Wv, wtv);
  transpose_w_kernel<<<dim3(1024), blk, 0, stream>>>(Wo, wto);
  // 2048 strips / 8 waves per block (32x64 per wave)
  proj_kernel<<<dim3(256), blk, 0, stream>>>(xb, wtq, bq, qws,  0);
  proj_kernel<<<dim3(256), blk, 0, stream>>>(xb, wtk, bk, kws,  0);
  proj_kernel<<<dim3(256), blk, 0, stream>>>(xb, wtv, bv, vtws, 1);
  // 4096 query tiles / 8 waves per block
  attention_kernel<<<dim3(512), blk, 0, stream>>>(qws, kws, vtws, aws);
  out_proj_kernel <<<dim3(256), blk, 0, stream>>>(aws, wto, bo, out);
}
